// Decoder_36283883716822
// MI455X (gfx1250) — matrix operand order repaired and checked by reading, not compiled, not run on hardware
//
#include <hip/hip_runtime.h>
#include <cstdint>
#include <cstddef>

#define DEV __device__ __forceinline__

typedef __bf16 bf16x4  __attribute__((ext_vector_type(4)));
typedef __bf16 bf16x8  __attribute__((ext_vector_type(8)));
typedef __bf16 bf16x16 __attribute__((ext_vector_type(16)));
typedef float  v8f     __attribute__((ext_vector_type(8)));

static constexpr int NN   = 4096; // graph nodes
static constexpr int BB   = 32;   // batch
static constexpr int HH   = 64;   // hidden
static constexpr int KK   = 3;    // Chebyshev hops
static constexpr int BM   = 128;  // output rows per workgroup
static constexpr int GSTR = 72;   // LDS stride for 64-wide K tiles (+8 bf16 pad -> bank spread)

// D = A(16x32 bf16) * B(32x16 bf16) + C(16x16 f32)
DEV v8f wmma_bf16(bf16x16 a, bf16x16 b, v8f c) {
  return __builtin_amdgcn_wmma_f32_16x16x32_bf16(false, a, false, b, (short)0, c, false, false);
}

// A-operand frag: lane lh=lane>>4 holds K = {8lh..8lh+7} and {16+8lh..16+8lh+7} of row (lane&15)
DEV bf16x16 ldA(const __bf16* base, int row, int kbase, int stride, int lh) {
  const __bf16* p = base + row * stride + kbase + 8 * lh;
  union { bf16x8 h[2]; bf16x16 v; } u;
  u.h[0] = *(const bf16x8*)(p);
  u.h[1] = *(const bf16x8*)(p + 16);
  return u.v;
}
// B-operand frag: lane lh=lane>>4 holds K = {8lh..8lh+7} and {16+8lh..16+8lh+7} for column (lane&15);
// storage is column-major ("row" index below is the matrix column).
DEV bf16x16 ldB(const __bf16* base, int col, int kbase, int stride, int lh) {
  const __bf16* p = base + col * stride + kbase + 8 * lh;
  union { bf16x8 h[2]; bf16x16 v; } u;
  u.h[0] = *(const bf16x8*)(p);
  u.h[1] = *(const bf16x8*)(p + 16);
  return u.v;
}

DEV float sigm(float x) { return 1.0f / (1.0f + __expf(-x)); }

// One-time f32 -> bf16 conversion of G so all 3 hops (100 MB) stay L2-resident (192 MB L2).
__global__ void g_cvt_bf16(const float* __restrict__ src, __bf16* __restrict__ dst, long n4) {
  long i      = (long)blockIdx.x * blockDim.x + threadIdx.x;
  long stride = (long)gridDim.x * blockDim.x;
  for (; i < n4; i += stride) {
    float4 v = ((const float4*)src)[i];
    bf16x4 o;
    o[0] = (__bf16)v.x; o[1] = (__bf16)v.y; o[2] = (__bf16)v.z; o[3] = (__bf16)v.w;
    ((bf16x4*)dst)[i] = o;
  }
}

// Fused layer kernel: for each (batch b, 128-row block):
//   loop k: S_k = G[k][rows,:] @ comb[b]  (WMMA, m-tiled through LDS)
//           gates += S_k @ W_k            (WMMA via LDS restage)
//   then bias + LSTM pointwise -> h_t, c_t
// P = CIN + HH (comb width), templated so accumulators stay in registers.
template <int P, int CIN, bool DUAL, bool GBF>
__global__ __launch_bounds__(256)
void glstm_layer(const void*  __restrict__ Gsrc,   // bf16 (GBF) or f32 G, [K][N][N]
                 const float* __restrict__ X,      // [B][N][CIN]  (x_t or h_t0)
                 const float* __restrict__ Hp,     // [B][N][HH]   (h_pre)
                 const float* __restrict__ Cpre,   // [B][N][HH]
                 const float* __restrict__ W,      // [K*P][256]
                 const float* __restrict__ bias,   // [256]
                 float* __restrict__ outH,
                 float* __restrict__ outC,
                 float* __restrict__ outH2)
{
  static_assert(P % 16 == 0, "P must be multiple of 16");
  constexpr int PT   = P / 16;               // supp column tiles per wave
  constexpr int PPAD = ((P + 31) / 32) * 32; // padded K-dim for the gate GEMM
  constexpr int KS2  = PPAD / 32;
  constexpr int WSTR = PPAD + 8;

  extern __shared__ __align__(16) char smem[];
  __bf16* Gt = (__bf16*)smem;        // [BM][GSTR]    G tile (rows x 64 k)
  __bf16* Ct = Gt + BM * GSTR;       // [P][GSTR]     comb^T tile (cols x 64 k)
  __bf16* Wt = Ct + P * GSTR;        // [256][WSTR]   W_k^T (zero-padded)
  __bf16* St = Wt + 256 * WSTR;      // [BM][WSTR]    staged S_k as bf16

  const int tid     = threadIdx.x;
  const int wave    = tid >> 5;      // 8 waves, 16 rows each
  const int lane    = tid & 31;
  const int l16     = lane & 15;
  const int lh      = lane >> 4;
  const int rowbase = blockIdx.x * BM;
  const int bI      = blockIdx.y;

  v8f gacc[16];                      // gates: 16 rows x 256 cols per wave
#pragma unroll
  for (int t = 0; t < 16; ++t) gacc[t] = {};

  for (int k = 0; k < KK; ++k) {
    __syncthreads();
    // ---- stage W_k^T (bf16), pad p in [P,PPAD) with zeros ----
    for (int idx = tid; idx < 256 * PPAD; idx += 256) {
      int p = idx >> 8;
      int j = idx & 255;
      float v = (p < P) ? W[(size_t)(k * P + p) * 256 + j] : 0.0f;
      Wt[j * WSTR + p] = (__bf16)v;
    }

    v8f sacc[PT];
#pragma unroll
    for (int pt = 0; pt < PT; ++pt) sacc[pt] = {};

    for (int m0 = 0; m0 < NN; m0 += 64) {
      __syncthreads();
      // ---- G tile: BM rows x 64 m-cols ----
#pragma unroll
      for (int it = 0; it < (BM * 64) / (256 * 8); ++it) {
        int idx = (it * 256 + tid) * 8;
        int r = idx >> 6;
        int c = idx & 63;
        size_t goff = (size_t)k * NN * NN + (size_t)(rowbase + r) * NN + m0 + c;
        __bf16* dst = &Gt[r * GSTR + c];
        if constexpr (GBF) {
          *(uint4*)dst = *(const uint4*)((const __bf16*)Gsrc + goff);
        } else {
          const float* s = (const float*)Gsrc + goff;
          float4 lo = *(const float4*)(s);
          float4 hi = *(const float4*)(s + 4);
          bf16x8 o;
          o[0] = (__bf16)lo.x; o[1] = (__bf16)lo.y; o[2] = (__bf16)lo.z; o[3] = (__bf16)lo.w;
          o[4] = (__bf16)hi.x; o[5] = (__bf16)hi.y; o[6] = (__bf16)hi.z; o[7] = (__bf16)hi.w;
          *(bf16x8*)dst = o;
        }
      }
      // ---- comb^T tile: P cols x 64 m (concat of X | Hp, converted to bf16) ----
      for (int idx = tid; idx < 64 * P; idx += 256) {
        int ml = idx / P;
        int p  = idx - ml * P;
        size_t row = (size_t)bI * NN + m0 + ml;
        float v = (p < CIN) ? X[row * CIN + p] : Hp[row * HH + (p - CIN)];
        Ct[p * GSTR + ml] = (__bf16)v;
      }
      __syncthreads();

      bf16x16 a0 = ldA(Gt, wave * 16 + l16, 0,  GSTR, lh);
      bf16x16 a1 = ldA(Gt, wave * 16 + l16, 32, GSTR, lh);
#pragma unroll
      for (int pt = 0; pt < PT; ++pt) {
        bf16x16 bA = ldB(Ct, pt * 16 + l16, 0,  GSTR, lh);
        sacc[pt] = wmma_bf16(a0, bA, sacc[pt]);
        bf16x16 bBv = ldB(Ct, pt * 16 + l16, 32, GSTR, lh);
        sacc[pt] = wmma_bf16(a1, bBv, sacc[pt]);
      }
    }

    __syncthreads();
    // ---- restage S_k as bf16 in A-operand-friendly layout; zero pad cols ----
    if constexpr (PPAD > P) {
      for (int idx = tid; idx < BM * (PPAD - P); idx += 256) {
        int r = idx / (PPAD - P);
        int p = P + (idx - r * (PPAD - P));
        St[r * WSTR + p] = (__bf16)0.0f;
      }
    }
#pragma unroll
    for (int pt = 0; pt < PT; ++pt) {
#pragma unroll
      for (int r = 0; r < 8; ++r) {
        // D frag element r of lane = (M = r + 8*lh, N = l16) within tile pt
        St[(wave * 16 + r + 8 * lh) * WSTR + pt * 16 + l16] = (__bf16)sacc[pt][r];
      }
    }
    __syncthreads();
    // ---- gates += S_k @ W_k ----
#pragma unroll
    for (int ks = 0; ks < KS2; ++ks) {
      bf16x16 a = ldA(St, wave * 16 + l16, ks * 32, WSTR, lh);
#pragma unroll
      for (int jt = 0; jt < 16; ++jt) {
        bf16x16 bw = ldB(Wt, jt * 16 + l16, ks * 32, WSTR, lh);
        gacc[jt] = wmma_bf16(a, bw, gacc[jt]);
      }
    }
  }

  // ---- bias + LSTM pointwise: gi=t0..3, gf=t4..7, go=t8..11, gg=t12..15 ----
#pragma unroll
  for (int t = 0; t < 4; ++t) {
    int h = t * 16 + l16;
    float bi = bias[h];
    float bf = bias[64 + h];
    float bo = bias[128 + h];
    float bg = bias[192 + h];
#pragma unroll
    for (int r = 0; r < 8; ++r) {
      int n = rowbase + wave * 16 + r + 8 * lh;
      size_t off = ((size_t)bI * NN + n) * HH + h;
      float gi = gacc[t][r]      + bi;
      float gf_ = gacc[t + 4][r] + bf;
      float go = gacc[t + 8][r]  + bo;
      float gg = gacc[t + 12][r] + bg;
      float cp = Cpre[off];
      float ct = sigm(gf_) * cp + sigm(gi) * tanhf(gg);
      float ht = sigm(go) * tanhf(ct);
      outC[off] = ct;
      outH[off] = ht;
      if constexpr (DUAL) outH2[off] = ht;
    }
  }
}

extern "C" void kernel_launch(void* const* d_in, const int* in_sizes, int n_in,
                              void* d_out, int out_size, void* d_ws, size_t ws_size,
                              hipStream_t stream) {
  (void)in_sizes; (void)n_in; (void)out_size;
  const float* G   = (const float*)d_in[0];
  const float* x_t = (const float*)d_in[1];
  const float* h0  = (const float*)d_in[2];
  const float* h1  = (const float*)d_in[3];
  const float* c0  = (const float*)d_in[4];
  const float* c1  = (const float*)d_in[5];
  const float* W0  = (const float*)d_in[6];
  const float* b0  = (const float*)d_in[7];
  const float* W1  = (const float*)d_in[8];
  const float* b1  = (const float*)d_in[9];

  float* out = (float*)d_out;
  const size_t S = (size_t)BB * NN * HH;
  float* h_t1_a = out;          // output tuple slot 0: h_t1
  float* h_t0   = out + S;      // slot 1: h_t0
  float* h_t1_b = out + 2 * S;  // slot 2: h_t1 (duplicate)
  float* c_t0   = out + 3 * S;  // slot 3: c_t0
  float* c_t1   = out + 4 * S;  // slot 4: c_t1

  const size_t gbytes = (size_t)KK * NN * NN * 2; // bf16 G in workspace
  const bool use_ws = (ws_size >= gbytes);

  if (use_ws) {
    long n4 = (long)KK * NN * NN / 4;
    g_cvt_bf16<<<4096, 256, 0, stream>>>(G, (__bf16*)d_ws, n4);
  }
  const void* Gsrc = use_ws ? (const void*)d_ws : (const void*)G;

  dim3 grid(NN / BM, BB); // (32, 32) workgroups of 256 threads (8 wave32s)
  constexpr int SH0 = (BM * GSTR + 80 * GSTR + 256 * 104 + BM * 104) * 2;  // ~110 KB LDS
  constexpr int SH1 = (BM * GSTR + 128 * GSTR + 256 * 136 + BM * 136) * 2; // ~141 KB LDS

  if (use_ws) {
    glstm_layer<80, 16, false, true><<<grid, 256, SH0, stream>>>(
        Gsrc, x_t, h0, c0, W0, b0, h_t0, c_t0, nullptr);
    glstm_layer<128, 64, true, true><<<grid, 256, SH1, stream>>>(
        Gsrc, h_t0, h1, c1, W1, b1, h_t1_a, c_t1, h_t1_b);
  } else {
    glstm_layer<80, 16, false, false><<<grid, 256, SH0, stream>>>(
        Gsrc, x_t, h0, c0, W0, b0, h_t0, c_t0, nullptr);
    glstm_layer<128, 64, true, false><<<grid, 256, SH1, stream>>>(
        Gsrc, h_t0, h1, c1, W1, b1, h_t1_a, c_t1, h_t1_b);
  }
}
